// AttentionBlock_23553600652048
// MI455X (gfx1250) — matrix operand order repaired and checked by reading, not compiled, not run on hardware
//
#include <hip/hip_runtime.h>
#include <hip/hip_bf16.h>

typedef __attribute__((ext_vector_type(16))) _Float16 v16h;
typedef __attribute__((ext_vector_type(8)))  _Float16 v8h;
typedef __attribute__((ext_vector_type(8)))  float    v8f;
typedef unsigned int v4u __attribute__((ext_vector_type(4)));
typedef int          v4i __attribute__((ext_vector_type(4)));
typedef int          v8i __attribute__((ext_vector_type(8)));

#define B_DIM 16
#define C_DIM 512
#define T_DIM 1024
#define H_DIM 8
#define DH    64
#define G_DIM 32

// ---------------------------------------------------------------------------
// Tensor Data Mover: stage a 2-D tile (t rows x c cols, f16) into LDS.
// D# packing per CDNA5 ISA ch.8: group0 = {flags/lds_addr/global_addr/type},
// group1 = {data_size, tensor dims, tile dims, dim0 stride}.
// ---------------------------------------------------------------------------
__device__ __forceinline__ void tdm_load_2d_f16(unsigned lds_byte_off,
                                                const void* gptr,
                                                unsigned tensor_d0, unsigned tensor_d1,
                                                unsigned tile_d0, unsigned tile_d1,
                                                unsigned stride_d0) {
  unsigned long long ga = (unsigned long long)(uintptr_t)gptr;
  v4u g0 = {0u, 0u, 0u, 0u};
  g0[0] = 1u;                                          // count=1, user descriptor
  g0[1] = lds_byte_off;                                // lds_addr [63:32]
  g0[2] = (unsigned)(ga & 0xFFFFFFFFu);                // global_addr [95:64]
  g0[3] = (unsigned)((ga >> 32) & 0x01FFFFFFu)         // global_addr [120:96]
        | (2u << 30);                                  // type=2 ("image")
  v8i g1 = {0, 0, 0, 0, 0, 0, 0, 0};
  g1[0] = (int)(1u << 16);                             // data_size=1 (2 bytes)
  g1[1] = (int)((tensor_d0 & 0xFFFFu) << 16);          // tensor_dim0 [79:48] lo
  g1[2] = (int)(((tensor_d0 >> 16) & 0xFFFFu)          // tensor_dim0 hi
        | ((tensor_d1 & 0xFFFFu) << 16));              // tensor_dim1 [111:80] lo
  g1[3] = (int)(((tensor_d1 >> 16) & 0xFFFFu)          // tensor_dim1 hi
        | ((tile_d0 & 0xFFFFu) << 16));                // tile_dim0 [127:112]
  g1[4] = (int)(tile_d1 & 0xFFFFu);                    // tile_dim1 [143:128]
  g1[5] = (int)stride_d0;                              // tensor_dim0_stride [207:160] lo32
  v4i z4 = {0, 0, 0, 0};
#if defined(__clang_major__) && (__clang_major__ >= 23)
  v8i z8 = {0, 0, 0, 0, 0, 0, 0, 0};
  __builtin_amdgcn_tensor_load_to_lds(g0, g1, z4, z4, z8, 0);
#else
  __builtin_amdgcn_tensor_load_to_lds(g0, g1, z4, z4, 0);
#endif
}

// ---------------------------------------------------------------------------
// WMMA fragment helpers (V_WMMA_F32_16X16X32_F16, wave32).
// A (16x32 MxK): lane m=lane&15; K segments [hi*8, hi*8+7] and [16+hi*8, ...].
// B (32x16 KxN): lane n=lane&15; load_frag_b K [hi*16, hi*16+15]; LOAD_FRAG_B_LDS same K order as A.
// C/D (16x16 f32): VGPR r -> rows r / r+8 by lane half, col = lane&15.
// ---------------------------------------------------------------------------
__device__ __forceinline__ v16h load_frag_a(const _Float16* __restrict__ src, int ld) {
  const int lane = threadIdx.x & 31;
  const int m  = lane & 15;
  const int hi = lane >> 4;
  const _Float16* p = src + (size_t)m * ld + hi * 8;
  v8h lo = *(const v8h*)(p);
  v8h up = *(const v8h*)(p + 16);
  v16h out;
#pragma unroll
  for (int j = 0; j < 8; ++j) { out[j] = lo[j]; out[8 + j] = up[j]; }
  return out;
}

// B fragment from N-major global storage: B[kk,n] = src[n*ld + kk]
__device__ __forceinline__ v16h load_frag_b(const _Float16* __restrict__ src, int ld) {
  const int lane = threadIdx.x & 31;
  const int n  = lane & 15;
  const int hi = lane >> 4;
  const _Float16* p = src + (size_t)n * ld + hi * 16;
  v8h lo = *(const v8h*)(p);
  v8h up = *(const v8h*)(p + 8);
  v16h out;
#pragma unroll
  for (int j = 0; j < 8; ++j) { out[j] = lo[j]; out[8 + j] = up[j]; }
  return out;
}

// B fragment from an LDS array (kept as a macro so addrspace(3) is preserved
// and the loads lower to ds_load_b128).
#define LOAD_FRAG_B_LDS(dst, arr, rowbase, ld, kbase)                          \
  {                                                                            \
    const int lane_ = threadIdx.x & 31;                                        \
    const int n_ = lane_ & 15;                                                 \
    const int hi_ = lane_ >> 4;                                                \
    v8h lo_ = *(const v8h*)&(arr)[((rowbase) + n_) * (ld) + (kbase) + hi_ * 8];      \
    v8h up_ = *(const v8h*)&(arr)[((rowbase) + n_) * (ld) + (kbase) + hi_ * 8 + 16];  \
    _Pragma("unroll") for (int j_ = 0; j_ < 8; ++j_) {                         \
      (dst)[j_] = lo_[j_]; (dst)[8 + j_] = up_[j_];                            \
    }                                                                          \
  }

__device__ __forceinline__ v8f wmma16(v16h a, v16h b, v8f c) {
  return __builtin_amdgcn_wmma_f32_16x16x32_f16(false, a, false, b, (short)0, c,
                                                false, false);
}

// ---------------------------------------------------------------------------
// f32 -> f16 weight conversion
// ---------------------------------------------------------------------------
__global__ void cvt_kernel(const float* __restrict__ src, _Float16* __restrict__ dst, int n) {
  int i = blockIdx.x * blockDim.x + threadIdx.x;
  if (i < n) dst[i] = (_Float16)src[i];
}

// ---------------------------------------------------------------------------
// GroupNorm: one workgroup per (batch, group); writes xnT in (b, t, c) f16
// ---------------------------------------------------------------------------
__global__ void gn_kernel(const float* __restrict__ x, const float* __restrict__ scale,
                          const float* __restrict__ bias, _Float16* __restrict__ xnT) {
  const int bb = blockIdx.x >> 5;
  const int g  = blockIdx.x & 31;
  const int cg = C_DIM / G_DIM;               // 16 channels per group
  const float* base = x + ((size_t)bb * C_DIM + (size_t)g * cg) * T_DIM;
  const int n = cg * T_DIM;                   // 16384
  float s = 0.f, s2 = 0.f;
  for (int i = threadIdx.x; i < n; i += blockDim.x) {
    float v = base[i];
    s += v; s2 += v * v;
  }
  __shared__ float red[256], red2[256];
  red[threadIdx.x] = s; red2[threadIdx.x] = s2;
  __syncthreads();
  for (int off = 128; off > 0; off >>= 1) {
    if ((int)threadIdx.x < off) {
      red[threadIdx.x]  += red[threadIdx.x + off];
      red2[threadIdx.x] += red2[threadIdx.x + off];
    }
    __syncthreads();
  }
  const float inv_n = 1.f / (float)n;
  const float mean = red[0] * inv_n;
  const float var  = red2[0] * inv_n - mean * mean;
  const float rstd = rsqrtf(var + 1e-5f);
  for (int i = threadIdx.x; i < n; i += blockDim.x) {
    int cl = i >> 10;
    int tt = i & 1023;
    int c  = g * cg + cl;
    float v = (base[i] - mean) * rstd * scale[c] + bias[c];
    xnT[((size_t)bb * T_DIM + tt) * C_DIM + c] = (_Float16)v;
  }
}

// ---------------------------------------------------------------------------
// QKV GEMM with TDM-staged B panel: one tensor_load_to_lds stages the
// 64 (t) x 512 (c) f16 strip (64 KB) for the whole workgroup; 8 waves then
// consume it from LDS (ds_load_b128) while streaming A (weights) from L2.
// Epilogue scatters to qT (bh,t,dh) [scaled dh^-0.5], kT (bh,t,dh), vB (bh,dh,t).
// ---------------------------------------------------------------------------
__global__ void qkv_kernel(const _Float16* __restrict__ w16, const float* __restrict__ bqkv,
                           const _Float16* __restrict__ xnT,
                           _Float16* __restrict__ qT, _Float16* __restrict__ kT,
                           _Float16* __restrict__ vB) {
  extern __shared__ _Float16 smem[];             // [64][C_DIM] = 64 KB
  const int wave = threadIdx.x >> 5;
  const int m0 = blockIdx.y * 128 + wave * 16;   // output-channel tile
  const int n0 = blockIdx.x * 64;                // t tile
  const int bb = blockIdx.z;

  if (threadIdx.x < 32) {                        // wave 0 drives the DMA
    tdm_load_2d_f16((unsigned)(uintptr_t)smem,
                    xnT + ((size_t)bb * T_DIM + n0) * C_DIM,
                    C_DIM, T_DIM, C_DIM, 64, C_DIM);
    __builtin_amdgcn_s_wait_tensorcnt(0);
  }
  __syncthreads();

  v8f acc[4] = {};
  const _Float16* abase = w16 + (size_t)m0 * C_DIM;
  for (int k0 = 0; k0 < C_DIM; k0 += 32) {
    __builtin_prefetch(abase + k0 + 32, 0, 1);
    v16h a = load_frag_a(abase + k0, C_DIM);
#pragma unroll
    for (int nt = 0; nt < 4; ++nt) {
      v16h bf;
      LOAD_FRAG_B_LDS(bf, smem, nt * 16, C_DIM, k0);
      acc[nt] = wmma16(a, bf, acc[nt]);
    }
  }
  const int lane = threadIdx.x & 31;
  const int col  = lane & 15;
  const int rb   = (lane >> 4) * 8;
#pragma unroll
  for (int nt = 0; nt < 4; ++nt) {
    int t_idx = n0 + nt * 16 + col;
#pragma unroll
    for (int r = 0; r < 8; ++r) {
      int o = m0 + r + rb;
      float val = acc[nt][r] + bqkv[o];
      int head = o / 192;
      int oi   = o - head * 192;
      int seg  = oi >> 6;
      int ch   = oi & 63;
      int bh   = bb * H_DIM + head;
      if (seg == 0)
        qT[((size_t)bh * T_DIM + t_idx) * DH + ch] = (_Float16)(val * 0.125f);
      else if (seg == 1)
        kT[((size_t)bh * T_DIM + t_idx) * DH + ch] = (_Float16)val;
      else
        vB[((size_t)bh * DH + ch) * T_DIM + t_idx] = (_Float16)val;
    }
  }
}

// ---------------------------------------------------------------------------
// Flash attention. The full K panel (1024x64) and V panel (64x1024) for this
// head-batch are TDM-loaded into 256 KB of LDS once, then the 32-step online-
// softmax loop runs entirely out of LDS + registers (8 WMMA per s-block).
// ---------------------------------------------------------------------------
__global__ void attn_kernel(const _Float16* __restrict__ qT, const _Float16* __restrict__ kT,
                            const _Float16* __restrict__ vB, _Float16* __restrict__ attnT) {
  extern __shared__ _Float16 smem[];             // kS [1024][64] | vS [64][1024]
  _Float16* kS = smem;                           // 128 KB
  _Float16* vS = smem + T_DIM * DH;              // 128 KB
  __shared__ _Float16 lds_p[8][16 * 32];         // per-wave P tile

  const int wave = threadIdx.x >> 5;
  const int lane = threadIdx.x & 31;
  const int col  = lane & 15;
  const int hi   = lane >> 4;
  const int rb   = hi * 8;
  const int bh   = blockIdx.y;
  const int t0   = blockIdx.x * 128 + wave * 16;

  if (threadIdx.x < 32) {
    tdm_load_2d_f16((unsigned)(uintptr_t)kS, kT + (size_t)bh * T_DIM * DH,
                    DH, T_DIM, DH, T_DIM, DH);           // 1024 rows of 64
    tdm_load_2d_f16((unsigned)(uintptr_t)vS, vB + (size_t)bh * DH * T_DIM,
                    T_DIM, DH, T_DIM, DH, T_DIM);        // 64 rows of 1024
    __builtin_amdgcn_s_wait_tensorcnt(0);
  }

  const v16h aq0 = load_frag_a(qT + ((size_t)bh * T_DIM + t0) * DH + 0,  DH);
  const v16h aq1 = load_frag_a(qT + ((size_t)bh * T_DIM + t0) * DH + 32, DH);
  __syncthreads();

  float m_i[8], l_i[8];
#pragma unroll
  for (int r = 0; r < 8; ++r) { m_i[r] = -1e30f; l_i[r] = 0.f; }
  v8f o_acc[4] = {};

  for (int s0 = 0; s0 < T_DIM; s0 += 32) {
    v8f c0 = {}, c1 = {};
    {
      v16h bk;
      LOAD_FRAG_B_LDS(bk, kS, s0,      DH, 0);  c0 = wmma16(aq0, bk, c0);
      LOAD_FRAG_B_LDS(bk, kS, s0,      DH, 32); c0 = wmma16(aq1, bk, c0);
      LOAD_FRAG_B_LDS(bk, kS, s0 + 16, DH, 0);  c1 = wmma16(aq0, bk, c1);
      LOAD_FRAG_B_LDS(bk, kS, s0 + 16, DH, 32); c1 = wmma16(aq1, bk, c1);
    }

#pragma unroll
    for (int r = 0; r < 8; ++r) {
      float bmax = fmaxf(c0[r], c1[r]);
#pragma unroll
      for (int off = 8; off > 0; off >>= 1)
        bmax = fmaxf(bmax, __shfl_xor(bmax, off, 32));   // reduce 16-lane half
      float mnew = fmaxf(m_i[r], bmax);
      float corr = __expf(m_i[r] - mnew);
      float p0 = __expf(c0[r] - mnew);
      float p1 = __expf(c1[r] - mnew);
      float ps = p0 + p1;
#pragma unroll
      for (int off = 8; off > 0; off >>= 1)
        ps += __shfl_xor(ps, off, 32);
      l_i[r] = l_i[r] * corr + ps;
      m_i[r] = mnew;
#pragma unroll
      for (int nt = 0; nt < 4; ++nt) o_acc[nt][r] *= corr;
      lds_p[wave][(r + rb) * 32 + col]      = (_Float16)p0;
      lds_p[wave][(r + rb) * 32 + col + 16] = (_Float16)p1;
    }

    // P back out of LDS in A-fragment layout
    v16h ap;
    {
      v8h lo = *(const v8h*)(&lds_p[wave][col * 32 + hi * 8]);
      v8h up = *(const v8h*)(&lds_p[wave][col * 32 + hi * 8 + 16]);
#pragma unroll
      for (int j = 0; j < 8; ++j) { ap[j] = lo[j]; ap[8 + j] = up[j]; }
    }
#pragma unroll
    for (int nt = 0; nt < 4; ++nt) {
      v16h bv;
      LOAD_FRAG_B_LDS(bv, vS, nt * 16, T_DIM, s0);
      o_acc[nt] = wmma16(ap, bv, o_acc[nt]);
    }
  }

  const int bbatch = bh >> 3;
  const int head   = bh & 7;
#pragma unroll
  for (int r = 0; r < 8; ++r) {
    float inv = 1.f / l_i[r];
    size_t row = (size_t)bbatch * T_DIM + t0 + r + rb;
#pragma unroll
    for (int nt = 0; nt < 4; ++nt)
      attnT[row * C_DIM + head * DH + nt * 16 + col] = (_Float16)(o_acc[nt][r] * inv);
  }
}

// ---------------------------------------------------------------------------
// Proj GEMM + bias + residual, TDM-staged B panel (attnT strip)
// ---------------------------------------------------------------------------
__global__ void proj_kernel(const _Float16* __restrict__ w16, const float* __restrict__ bproj,
                            const _Float16* __restrict__ attnT, const float* __restrict__ x,
                            float* __restrict__ out) {
  extern __shared__ _Float16 smem[];             // [64][C_DIM] = 64 KB
  const int wave = threadIdx.x >> 5;
  const int m0 = blockIdx.y * 128 + wave * 16;
  const int n0 = blockIdx.x * 64;
  const int bb = blockIdx.z;

  if (threadIdx.x < 32) {
    tdm_load_2d_f16((unsigned)(uintptr_t)smem,
                    attnT + ((size_t)bb * T_DIM + n0) * C_DIM,
                    C_DIM, T_DIM, C_DIM, 64, C_DIM);
    __builtin_amdgcn_s_wait_tensorcnt(0);
  }
  __syncthreads();

  v8f acc[4] = {};
  const _Float16* abase = w16 + (size_t)m0 * C_DIM;
  for (int k0 = 0; k0 < C_DIM; k0 += 32) {
    __builtin_prefetch(abase + k0 + 32, 0, 1);
    v16h a = load_frag_a(abase + k0, C_DIM);
#pragma unroll
    for (int nt = 0; nt < 4; ++nt) {
      v16h bf;
      LOAD_FRAG_B_LDS(bf, smem, nt * 16, C_DIM, k0);
      acc[nt] = wmma16(a, bf, acc[nt]);
    }
  }
  const int lane = threadIdx.x & 31;
  const int col  = lane & 15;
  const int rb   = (lane >> 4) * 8;
#pragma unroll
  for (int nt = 0; nt < 4; ++nt) {
    int t_idx = n0 + nt * 16 + col;
#pragma unroll
    for (int r = 0; r < 8; ++r) {
      int o = m0 + r + rb;
      size_t idx = ((size_t)bb * C_DIM + o) * T_DIM + t_idx;
      out[idx] = x[idx] + bproj[o] + acc[nt][r];
    }
  }
}

// ---------------------------------------------------------------------------
extern "C" void kernel_launch(void* const* d_in, const int* in_sizes, int n_in,
                              void* d_out, int out_size, void* d_ws, size_t ws_size,
                              hipStream_t stream) {
  (void)in_sizes; (void)n_in; (void)out_size; (void)ws_size;
  const float* x        = (const float*)d_in[0];
  const float* gn_scale = (const float*)d_in[1];
  const float* gn_bias  = (const float*)d_in[2];
  const float* w_qkv    = (const float*)d_in[3];
  const float* b_qkv    = (const float*)d_in[4];
  const float* w_proj   = (const float*)d_in[5];
  const float* b_proj   = (const float*)d_in[6];
  float* out = (float*)d_out;

  char* ws = (char*)d_ws;
  size_t off = 0;
  const size_t btc = (size_t)B_DIM * T_DIM * C_DIM * sizeof(_Float16);   // 16 MB
  _Float16* xnT   = (_Float16*)(ws + off); off += btc;                   // (b,t,c)
  _Float16* qT    = (_Float16*)(ws + off); off += btc;                   // (bh,t,dh)
  _Float16* kT    = (_Float16*)(ws + off); off += btc;                   // (bh,t,dh)
  _Float16* vB    = (_Float16*)(ws + off); off += btc;                   // (bh,dh,t)
  _Float16* attnT = (_Float16*)(ws + off); off += btc;                   // (b,t,c)
  _Float16* wq16  = (_Float16*)(ws + off); off += (size_t)3 * C_DIM * C_DIM * sizeof(_Float16);
  _Float16* wp16  = (_Float16*)(ws + off); off += (size_t)C_DIM * C_DIM * sizeof(_Float16);

  static bool attr_done = false;   // host-side launch config, not device state
  if (!attr_done) {
    hipFuncSetAttribute((const void*)attn_kernel,
                        hipFuncAttributeMaxDynamicSharedMemorySize, 262144);
    hipFuncSetAttribute((const void*)qkv_kernel,
                        hipFuncAttributeMaxDynamicSharedMemorySize, 65536);
    hipFuncSetAttribute((const void*)proj_kernel,
                        hipFuncAttributeMaxDynamicSharedMemorySize, 65536);
    attr_done = true;
  }

  const int nq = 3 * C_DIM * C_DIM, np = C_DIM * C_DIM;
  cvt_kernel<<<(nq + 255) / 256, 256, 0, stream>>>(w_qkv, wq16, nq);
  cvt_kernel<<<(np + 255) / 256, 256, 0, stream>>>(w_proj, wp16, np);

  gn_kernel<<<B_DIM * G_DIM, 256, 0, stream>>>(x, gn_scale, gn_bias, xnT);

  // M=1536 (12 blocks of 128), N=1024 (16 blocks of 64), 16 batches
  qkv_kernel<<<dim3(16, 12, 16), 256, 65536, stream>>>(wq16, b_qkv, xnT, qT, kT, vB);

  // 8 row-blocks of 128 per head-batch, 128 head-batches
  attn_kernel<<<dim3(8, 128), 256, 262144, stream>>>(qT, kT, vB, attnT);

  // M=512 (4 blocks of 128), N=1024 (16 blocks of 64), 16 batches
  proj_kernel<<<dim3(16, 4, 16), 256, 65536, stream>>>(wp16, b_proj, attnT, x, out);
}
